// RBNN_68470368633577
// MI455X (gfx1250) — matrix operand order repaired and checked by reading, not compiled, not run on hardware
//
#include <hip/hip_runtime.h>
#include <hip/hip_bf16.h>

// ---------------- problem constants (from reference) ----------------
#define B_    64
#define T_    2000
#define IN_   128
#define HID_  512
#define OUT_  128
#define RING  6          // DELAY + 1
#define DECAY 0.999f     // 1 - DT*KM
#define GAIN  1.0e-4f    // DT*KM*R

// ---------------- kernel geometry ----------------
#define BT      16                 // batch rows per workgroup (WMMA M tile)
#define NWG     (B_ / BT)          // 4 persistent workgroups, no cross-WG sync
#define WAVES   8
#define THREADS (WAVES * 32)

#define SROW  520                  // padded LDS row stride (halves) for s ring: 1040B, 16B aligned
#define XROWF 132                  // padded LDS row stride (floats) for f32 x tile: 528B, 16B aligned
#define KT_HID (HID_ / 32)         // 16 K-fragments over HID
#define KT_IN  (IN_  / 32)         // 4  K-fragments over IN

// LDS layout (dynamic shared segment starts at LDS offset 0: no static LDS here)
#define RING_HALVES (RING * BT * SROW)          // 49920 halves = 99840 B
#define XBUF_FLOATS (BT * XROWF)                // one f32 x tile (padded)
#define XF_BASE     (RING_HALVES * 2)           // byte offset of x double-buffer
#define SMEM_BYTES  (XF_BASE + 2 * XBUF_FLOATS * 4)  // ~114 KB

typedef __attribute__((ext_vector_type(16))) _Float16 v16h;
typedef __attribute__((ext_vector_type(8)))  _Float16 v8h;
typedef __attribute__((ext_vector_type(8)))  float    v8f;
typedef __attribute__((ext_vector_type(4)))  float    v4f;

// --------------------------------------------------------------------
// Pre-swizzle a weight matrix W[N][K] (f32, row major) into WMMA
// B-fragment order (f16): fragment (tn, tk) holds B[k][n] = W[tn*16+n][tk*32+k].
// CDNA5 16-bit B 32x16 layout: lane L holds n = L&15, K = (e>>3)*16 + (L>>4)*8 + (e&7),
// 16 contiguous halves per lane -> one 32B load per lane in the main kernel.
// --------------------------------------------------------------------
__global__ void swizzle_weights(const float* __restrict__ src,
                                _Float16* __restrict__ dst, int N, int K) {
  int i = blockIdx.x * blockDim.x + threadIdx.x;
  if (i >= N * K) return;
  int e      = i & 15;
  int lane   = (i >> 4) & 31;
  int frag   = i >> 9;
  int ktiles = K >> 5;
  int tn = frag / ktiles;
  int tk = frag - tn * ktiles;
  int row = tn * 16 + (lane & 15);
  int col = tk * 32 + ((e >> 3) << 4) + ((lane >> 4) << 3) + (e & 7);
  dst[i] = (_Float16)src[row * K + col];
}

// helper: issue async global->LDS copy of one x_t tile into buffer `buf`
__device__ __forceinline__ void stage_x_async(const float* __restrict__ x,
                                              int b0, int t, int buf, int tid) {
#pragma unroll
  for (int c = 0; c < 2; ++c) {
    int chunk = tid + c * THREADS;          // 0..511, 16B each
    int row   = chunk >> 5;                 // batch row 0..15
    int col16 = chunk & 31;                 // 16B column chunk
    unsigned lds_off =
        (unsigned)(XF_BASE + buf * (XBUF_FLOATS * 4) + row * (XROWF * 4) + col16 * 16);
    const float* gsrc =
        x + (size_t)(b0 + row) * T_ * IN_ + (size_t)t * IN_ + col16 * 4;
    asm volatile("global_load_async_to_lds_b128 %0, %1, off"
                 :: "v"(lds_off), "v"(gsrc)
                 : "memory");
  }
}

// load a 16x32 f16 A-fragment from an LDS row-major tile (stride in halves)
__device__ __forceinline__ v16h load_a_f16(const _Float16* base, int arow,
                                           int stride, int kf, int h) {
  v8h lo = *(const v8h*)(base + arow * stride + kf * 32 + h * 8);
  v8h hi = *(const v8h*)(base + arow * stride + kf * 32 + 16 + h * 8);
  return __builtin_shufflevector(lo, hi, 0, 1, 2, 3, 4, 5, 6, 7, 8, 9, 10, 11,
                                 12, 13, 14, 15);
}

// --------------------------------------------------------------------
// Persistent GLIFR step kernel. Each workgroup owns 16 batch rows for all
// 2000 steps. Membrane voltage v lives in WMMA C-layout accumulator
// registers (4 hid tiles per wave). Firing history is a 6-slot LDS ring.
// Software pipeline per step t:
//   top:    wait async x_t; issue async x_{t+1}; pre = rec(t) + x_t@Win^T;
//           membrane update; sigmoid; write s_t
//   bottom: out-GEMM(t), then rec-GEMM for t+1 (reads s_{t-4}, legal thanks
//           to DELAY=5) as two register-light loops; result carried to t+1.
// --------------------------------------------------------------------
__launch_bounds__(THREADS)
__global__ void rbnn_persistent(const float* __restrict__ x,
                                const float* __restrict__ b_in,
                                const float* __restrict__ b_rec,
                                const float* __restrict__ b_out,
                                const _Float16* __restrict__ Wrec_f,  // 32x16 frags
                                const _Float16* __restrict__ Win_f,   // 32x4  frags
                                const _Float16* __restrict__ Wout_f,  // 8x16  frags
                                float* __restrict__ out) {
  extern __shared__ _Float16 smem[];
  _Float16* s_ring = smem;                                   // ring of s tiles (f16)
  float*    s_xf   = (float*)(smem + RING_HALVES);           // f32 x double buffer

  const int tid  = threadIdx.x;
  const int lane = tid & 31;
  const int w    = tid >> 5;          // wave id 0..7
  const int b0   = blockIdx.x * BT;   // batch row base

  // zero the firing-history ring (hist0 == zeros in the reference)
  for (int i = tid; i < RING_HALVES; i += THREADS) s_ring[i] = (_Float16)0.0f;
  __syncthreads();

  const int arow = lane & 15;         // A-fragment row (M) for this lane
  const int h    = lane >> 4;         // A-fragment K-half selector
  const int mrow = h << 3;            // C-layout: this lane holds M = mrow + r
  const int ncol = lane & 15;         // C-layout: this lane holds N = ncol

  // persistent state: membrane voltage + carried recurrent-GEMM result
  v8f   vmem[4] = {};
  v8f   rec[4]  = {};                 // rec-drive for step t (zero: s_{<0} == 0)
  float biasPre[4];
#pragma unroll
  for (int j = 0; j < 4; ++j) {
    int n = (w * 4 + j) * 16 + ncol;
    biasPre[j] = b_in[n] + b_rec[n];
  }
  const float biasOut = b_out[w * 16 + ncol];

  // prologue: start async staging of x_0 into buffer 0
  stage_x_async(x, b0, 0, 0, tid);

  for (int t = 0; t < T_; ++t) {
    const int p = t & 1;
    // ---- wait for x_t tile, publish; then start staging x_{t+1} ----
    asm volatile("s_wait_asynccnt 0x0" ::: "memory");
    __syncthreads();
    {
      int tn = (t + 1 < T_) ? t + 1 : T_ - 1;   // clamp: avoid OOB prefetch
      stage_x_async(x, b0, tn, 1 - p, tid);
    }
    const float* xbuf = s_xf + p * XBUF_FLOATS;

    // ---- build all 4 input A-fragments first (f32 LDS -> f16 regs), so the
    //      in-GEMM below is a pure B-fragment global stream ----
    v16h ax[KT_IN];
#pragma unroll
    for (int kf = 0; kf < KT_IN; ++kf) {
      const float* xr = xbuf + arow * XROWF + kf * 32 + h * 8;
      v4f f0 = *(const v4f*)(xr);
      v4f f1 = *(const v4f*)(xr + 4);
      v4f f2 = *(const v4f*)(xr + 16);
      v4f f3 = *(const v4f*)(xr + 20);
#pragma unroll
      for (int i2 = 0; i2 < 4; ++i2) {
        ax[kf][i2]      = (_Float16)f0[i2];
        ax[kf][i2 + 4]  = (_Float16)f1[i2];
        ax[kf][i2 + 8]  = (_Float16)f2[i2];
        ax[kf][i2 + 12] = (_Float16)f3[i2];
      }
    }

    // ---- pre = rec (carried) + x_t @ W_in^T ----
    v8f pre[4];
#pragma unroll
    for (int j = 0; j < 4; ++j) pre[j] = rec[j];
#pragma unroll
    for (int kf = 0; kf < KT_IN; ++kf) {
#pragma unroll
      for (int j = 0; j < 4; ++j) {
        const v16h bf =
            *(const v16h*)(Win_f + (((size_t)(w * 4 + j) * KT_IN + kf) * 32 + lane) * 16);
        pre[j] = __builtin_amdgcn_wmma_f32_16x16x32_f16(false, ax[kf], false, bf,
                                                        (short)0, pre[j], false, false);
      }
    }

    // ---- membrane update + sigmoid, write s_t into ring slot t%RING ----
    _Float16* ringW = s_ring + (t % RING) * BT * SROW;
#pragma unroll
    for (int j = 0; j < 4; ++j) {
      int n = (w * 4 + j) * 16 + ncol;
#pragma unroll
      for (int r = 0; r < 8; ++r) {
        float v = vmem[j][r] * DECAY + GAIN * (pre[j][r] + biasPre[j]);
        vmem[j][r] = v;
        float s = __builtin_amdgcn_rcpf(1.0f + __expf(-v));
        ringW[(mrow + r) * SROW + n] = (_Float16)s;
      }
    }
    __syncthreads();

    // ---- out-GEMM(t): out_t = s_t @ W_out^T + b_out (wave w -> out tile w) ----
    v8f acc = {};
    for (int kf = 0; kf < KT_HID; ++kf) {
      v16h aO = load_a_f16(ringW, arow, SROW, kf, h);
      const v16h bO =
          *(const v16h*)(Wout_f + (((size_t)w * KT_HID + kf) * 32 + lane) * 16);
      acc = __builtin_amdgcn_wmma_f32_16x16x32_f16(false, aO, false, bO, (short)0,
                                                   acc, false, false);
    }
#pragma unroll
    for (int r = 0; r < 8; ++r) {
      int m = mrow + r;
      out[(size_t)(b0 + m) * T_ * OUT_ + (size_t)t * OUT_ + (w * 16 + ncol)] =
          acc[r] + biasOut;
    }

    // ---- rec-GEMM for t+1: rec = s_{t-4} @ W_rec^T
    //      (ring slot (t+2)%RING: written at t-4, next write at t+2 -> safe) ----
    const _Float16* ringN = s_ring + ((t + 2) % RING) * BT * SROW;
#pragma unroll
    for (int j = 0; j < 4; ++j) rec[j] = (v8f){};
    for (int kf = 0; kf < KT_HID; ++kf) {
      v16h aR = load_a_f16(ringN, arow, SROW, kf, h);
      __builtin_prefetch(
          Wrec_f + (((size_t)(w * 4) * KT_HID + ((kf + 1) & (KT_HID - 1))) * 32 + lane) * 16,
          0, 1);
#pragma unroll
      for (int j = 0; j < 4; ++j) {
        const v16h bf =
            *(const v16h*)(Wrec_f + (((size_t)(w * 4 + j) * KT_HID + kf) * 32 + lane) * 16);
        rec[j] = __builtin_amdgcn_wmma_f32_16x16x32_f16(false, aR, false, bf,
                                                        (short)0, rec[j], false, false);
      }
    }
    // no trailing barrier: next top's async writes go to the other x buffer and
    // are only made visible after its own s_wait_asynccnt + barrier; ring slot
    // (t+1)%RING is written only after that barrier too.
  }
}

// --------------------------------------------------------------------
extern "C" void kernel_launch(void* const* d_in, const int* in_sizes, int n_in,
                              void* d_out, int out_size, void* d_ws, size_t ws_size,
                              hipStream_t stream) {
  const float* x     = (const float*)d_in[0];
  const float* W_in  = (const float*)d_in[1];
  const float* b_in  = (const float*)d_in[2];
  const float* W_rec = (const float*)d_in[3];
  const float* b_rec = (const float*)d_in[4];
  const float* W_out = (const float*)d_in[5];
  const float* b_out = (const float*)d_in[6];
  float* out = (float*)d_out;

  // workspace: f16 fragment-swizzled weights (768 KB)
  _Float16* Wrec_f = (_Float16*)d_ws;                 // HID*HID
  _Float16* Win_f  = Wrec_f + (size_t)HID_ * HID_;    // HID*IN
  _Float16* Wout_f = Win_f + (size_t)HID_ * IN_;      // OUT*HID

  swizzle_weights<<<(HID_ * HID_ + 255) / 256, 256, 0, stream>>>(W_rec, Wrec_f, HID_, HID_);
  swizzle_weights<<<(HID_ * IN_ + 255) / 256, 256, 0, stream>>>(W_in, Win_f, HID_, IN_);
  swizzle_weights<<<(OUT_ * HID_ + 255) / 256, 256, 0, stream>>>(W_out, Wout_f, OUT_, HID_);

  rbnn_persistent<<<NWG, THREADS, SMEM_BYTES, stream>>>(x, b_in, b_rec, b_out,
                                                        Wrec_f, Win_f, Wout_f, out);
}
